// Attention_55319178772570
// MI455X (gfx1250) — matrix operand order repaired and checked by reading, not compiled, not run on hardware
//
#include <hip/hip_runtime.h>

typedef __attribute__((ext_vector_type(16))) _Float16 v16h;
typedef __attribute__((ext_vector_type(8)))  _Float16 v8h;
typedef __attribute__((ext_vector_type(8)))  float    v8f;

namespace {

constexpr int kB   = 16;
constexpr int kL   = 512;
constexpr int kD   = 768;
constexpr int kH   = 12;
constexpr int kDH  = 64;
constexpr int kM   = kB * kL;     // 8192 tokens
constexpr int kNQ  = 3 * kD;      // 2304 fused QKV output columns

__device__ __forceinline__ v16h cat8(v8h lo, v8h hi) {
  return __builtin_shufflevector(lo, hi, 0, 1, 2, 3, 4, 5, 6, 7,
                                 8, 9, 10, 11, 12, 13, 14, 15);
}

__device__ __forceinline__ v8f wmma16(v16h a, v16h b, v8f c) {
  // D = A(16x32 f16) * B(32x16 f16) + C(16x16 f32)
  return __builtin_amdgcn_wmma_f32_16x16x32_f16(false, a, false, b, (short)0, c,
                                                false, false);
}

// ---------------- prep: fp32 -> f16 conversions / weight transposes ----------
__global__ void k_cvt_x(const float* __restrict__ x, _Float16* __restrict__ xh) {
  size_t i = (size_t)blockIdx.x * blockDim.x + threadIdx.x;
  if (i < (size_t)kM * kD) xh[i] = (_Float16)x[i];
}

__global__ void k_prep_wqkv(const float* __restrict__ Wq, const float* __restrict__ Wk,
                            const float* __restrict__ Wv, const float* __restrict__ bq,
                            const float* __restrict__ bk, const float* __restrict__ bv,
                            _Float16* __restrict__ wt, float* __restrict__ bias) {
  size_t i = (size_t)blockIdx.x * blockDim.x + threadIdx.x;
  if (i >= (size_t)kNQ * kD) return;
  int n = (int)(i / kD), k = (int)(i % kD);
  int which = n / kD, nn = n % kD;
  const float* W = (which == 0) ? Wq : (which == 1) ? Wk : Wv;
  wt[(size_t)n * kD + k] = (_Float16)W[(size_t)k * kD + nn];  // wt = W^T
  if (k == 0) {
    const float* bb = (which == 0) ? bq : (which == 1) ? bk : bv;
    bias[n] = bb[nn];
  }
}

__global__ void k_prep_wo(const float* __restrict__ Wo, _Float16* __restrict__ wt) {
  size_t i = (size_t)blockIdx.x * blockDim.x + threadIdx.x;
  if (i >= (size_t)kD * kD) return;
  int n = (int)(i / kD), k = (int)(i % kD);
  wt[(size_t)n * kD + k] = (_Float16)Wo[(size_t)k * kD + n];
}

// ---------------- fused QKV projection GEMM: [8192x768] x [768x2304] ---------
// Per wave: 32(M) x 64(N) tile; WG = 8 waves -> 128x128; grid (64, 18).
// Epilogue scatters Q,K row-major [B,H,L,DH] and V transposed [B,H,DH,L].
__global__ __launch_bounds__(256) void k_gemm_qkv(
    const _Float16* __restrict__ xh, const _Float16* __restrict__ wt,
    const float* __restrict__ bias, _Float16* __restrict__ qb,
    _Float16* __restrict__ kb, _Float16* __restrict__ vtb) {
  const int lane = threadIdx.x & 31, wave = threadIdx.x >> 5;
  const int lane15 = lane & 15, half = lane >> 4;
  const int m0 = blockIdx.x * 128 + (wave & 3) * 32;
  const int n0 = blockIdx.y * 128 + (wave >> 2) * 64;

  v8f zero = {};
  v8f acc[2][4];
#pragma unroll
  for (int mi = 0; mi < 2; ++mi)
#pragma unroll
    for (int ni = 0; ni < 4; ++ni) acc[mi][ni] = zero;

  for (int k0 = 0; k0 < kD; k0 += 32) {
    v16h af[2];
#pragma unroll
    for (int mi = 0; mi < 2; ++mi) {
      const _Float16* ap = xh + (size_t)(m0 + mi * 16 + lane15) * kD + k0 + 8 * half;
      af[mi] = cat8(*(const v8h*)ap, *(const v8h*)(ap + 16));
      __builtin_prefetch(ap + 32, 0, 3);  // speculative; next K-slab
    }
    v16h bf[4];
#pragma unroll
    for (int ni = 0; ni < 4; ++ni) {
      const _Float16* bp = wt + (size_t)(n0 + ni * 16 + lane15) * kD + k0 + 8 * half;
      bf[ni] = cat8(*(const v8h*)bp, *(const v8h*)(bp + 16));
      __builtin_prefetch(bp + 32, 0, 3);
    }
#pragma unroll
    for (int mi = 0; mi < 2; ++mi)
#pragma unroll
      for (int ni = 0; ni < 4; ++ni) acc[mi][ni] = wmma16(af[mi], bf[ni], acc[mi][ni]);
  }

#pragma unroll
  for (int ni = 0; ni < 4; ++ni) {
    const int n = n0 + ni * 16 + lane15;
    const float bn = bias[n];
    const int which = n / kD, nn = n % kD;
    const int h = nn >> 6, d = nn & 63;
#pragma unroll
    for (int mi = 0; mi < 2; ++mi) {
#pragma unroll
      for (int r = 0; r < 8; ++r) {
        const int m = m0 + mi * 16 + r + 8 * half;
        const int b = m >> 9, l = m & 511;
        const _Float16 v = (_Float16)(acc[mi][ni][r] + bn);
        if (which == 0)
          qb[(((size_t)b * kH + h) * kL + l) * kDH + d] = v;
        else if (which == 1)
          kb[(((size_t)b * kH + h) * kL + l) * kDH + d] = v;
        else
          vtb[(((size_t)b * kH + h) * kDH + d) * kL + l] = v;  // V transposed
      }
    }
  }
}

// ---------------- flash attention: one WG = (b, h, 128-query block) ----------
// Each wave owns 16 queries; streams keys in blocks of 32.
// Computes S^T = K*Q^T so softmax is lane-local and P^T -> A-fragment is free.
__global__ __launch_bounds__(256) void k_attn(
    const _Float16* __restrict__ qb, const _Float16* __restrict__ kb,
    const _Float16* __restrict__ vtb, const int* __restrict__ mask,
    _Float16* __restrict__ ctx) {
  const int lane = threadIdx.x & 31, wave = threadIdx.x >> 5;
  const int lane15 = lane & 15, half = lane >> 4;
  const int qblk = blockIdx.x & 3, bh = blockIdx.x >> 2;
  const int b = bh / kH, h = bh % kH;
  const int qbase = qblk * 128 + wave * 16;
  const int q = qbase + lane15;

  // Q B-fragments for the two d-steps of S^T; resident across all key blocks.
  const _Float16* qp = qb + ((size_t)bh * kL + q) * kDH;
  v16h qf[2];
  qf[0] = cat8(*(const v8h*)(qp + 8 * half), *(const v8h*)(qp + 16 + 8 * half));
  qf[1] = cat8(*(const v8h*)(qp + 32 + 8 * half), *(const v8h*)(qp + 48 + 8 * half));

  v8f zero = {};
  v8f o[4];
#pragma unroll
  for (int t = 0; t < 4; ++t) o[t] = zero;
  float m_run = -1e30f, l_run = 0.0f;

  for (int kblk = 0; kblk < kL / 32; ++kblk) {
    const int key0 = kblk * 32;
    // key-side padding mask: lane <-> key (key0 + lane)
    const float sbias = (mask[(size_t)b * kL + key0 + lane] > 0) ? 0.0f : -1e9f;

    // S^T tiles: M = 32 keys (2 tiles), N = 16 queries, KK = 64 (2 steps of 32)
    v8f s[2];
    s[0] = zero; s[1] = zero;
#pragma unroll
    for (int ds = 0; ds < 2; ++ds) {
#pragma unroll
      for (int kt = 0; kt < 2; ++kt) {
        const _Float16* kp = kb + ((size_t)bh * kL + key0 + kt * 16 + lane15) * kDH +
                             ds * 32 + 8 * half;
        v16h ka = cat8(*(const v8h*)kp, *(const v8h*)(kp + 16));
        s[kt] = wmma16(ka, qf[ds], s[kt]);
      }
    }

    // scale + mask; block max over this lane's 16 keys, then combine halves
    float bm = -1e30f;
#pragma unroll
    for (int r = 0; r < 8; ++r) {
      const float mb0 = __shfl(sbias, r + 8 * half);
      const float mb1 = __shfl(sbias, 16 + r + 8 * half);
      s[0][r] = s[0][r] * 0.125f + mb0;   // SCALE = 1/sqrt(64)
      s[1][r] = s[1][r] * 0.125f + mb1;
      bm = fmaxf(bm, fmaxf(s[0][r], s[1][r]));
    }
    bm = fmaxf(bm, __shfl_xor(bm, 16));
    const float m_new = fmaxf(m_run, bm);
    const float alpha = __expf(m_run - m_new);

    float lsum = 0.0f;
    v16h pa;  // P^T tile folded straight into A-fragment layout
#pragma unroll
    for (int r = 0; r < 8; ++r) {
      const float e0 = __expf(s[0][r] - m_new);
      const float e1 = __expf(s[1][r] - m_new);
      lsum += e0 + e1;
      pa[r] = (_Float16)e0;
      pa[r + 8] = (_Float16)e1;
    }
    lsum += __shfl_xor(lsum, 16);
    l_run = l_run * alpha + lsum;
    m_run = m_new;

    // rescale running O (rows = queries; broadcast alpha per row)
#pragma unroll
    for (int r = 0; r < 8; ++r) {
      const float ar = __shfl(alpha, r + 8 * half);
#pragma unroll
      for (int t = 0; t < 4; ++t) o[t][r] *= ar;
    }

    // O += P * V : N = 64 dims (4 tiles), KK = 32 keys (V^T gives contiguous B-frags)
    const _Float16* vp = vtb + ((size_t)bh * kDH + lane15) * kL + key0 + 8 * half;
#pragma unroll
    for (int t = 0; t < 4; ++t) {
      v16h vf = cat8(*(const v8h*)(vp + (size_t)t * 16 * kL), *(const v8h*)(vp + (size_t)t * 16 * kL + 16));
      o[t] = wmma16(pa, vf, o[t]);
    }
  }

  // normalize and scatter ctx[token][h*64 + d] as f16 for the out-proj GEMM
#pragma unroll
  for (int r = 0; r < 8; ++r) {
    const float inv = 1.0f / __shfl(l_run, r + 8 * half);
    const int ql = qbase + r + 8 * half;
    _Float16* cp = ctx + ((size_t)b * kL + ql) * kD + h * kDH + lane15;
#pragma unroll
    for (int t = 0; t < 4; ++t) cp[t * 16] = (_Float16)(o[t][r] * inv);
  }
}

// ---------------- out-projection GEMM + bias + residual ----------------------
__global__ __launch_bounds__(256) void k_gemm_out(
    const _Float16* __restrict__ ctx, const _Float16* __restrict__ wt,
    const float* __restrict__ bo, const float* __restrict__ x,
    float* __restrict__ y) {
  const int lane = threadIdx.x & 31, wave = threadIdx.x >> 5;
  const int lane15 = lane & 15, half = lane >> 4;
  const int m0 = blockIdx.x * 128 + (wave & 3) * 32;
  const int n0 = blockIdx.y * 128 + (wave >> 2) * 64;

  v8f zero = {};
  v8f acc[2][4];
#pragma unroll
  for (int mi = 0; mi < 2; ++mi)
#pragma unroll
    for (int ni = 0; ni < 4; ++ni) acc[mi][ni] = zero;

  for (int k0 = 0; k0 < kD; k0 += 32) {
    v16h af[2];
#pragma unroll
    for (int mi = 0; mi < 2; ++mi) {
      const _Float16* ap = ctx + (size_t)(m0 + mi * 16 + lane15) * kD + k0 + 8 * half;
      af[mi] = cat8(*(const v8h*)ap, *(const v8h*)(ap + 16));
      __builtin_prefetch(ap + 32, 0, 3);
    }
    v16h bf[4];
#pragma unroll
    for (int ni = 0; ni < 4; ++ni) {
      const _Float16* bp = wt + (size_t)(n0 + ni * 16 + lane15) * kD + k0 + 8 * half;
      bf[ni] = cat8(*(const v8h*)bp, *(const v8h*)(bp + 16));
      __builtin_prefetch(bp + 32, 0, 3);
    }
#pragma unroll
    for (int mi = 0; mi < 2; ++mi)
#pragma unroll
      for (int ni = 0; ni < 4; ++ni) acc[mi][ni] = wmma16(af[mi], bf[ni], acc[mi][ni]);
  }

#pragma unroll
  for (int ni = 0; ni < 4; ++ni) {
    const int n = n0 + ni * 16 + lane15;
    const float bn = bo[n];
#pragma unroll
    for (int mi = 0; mi < 2; ++mi) {
#pragma unroll
      for (int r = 0; r < 8; ++r) {
        const int m = m0 + mi * 16 + r + 8 * half;
        y[(size_t)m * kD + n] = acc[mi][ni][r] + bn + x[(size_t)m * kD + n];
      }
    }
  }
}

// ---------------- LayerNorm: one wave per row (768 cols, 24/lane) ------------
__global__ __launch_bounds__(256) void k_layernorm(
    const float* __restrict__ y, const float* __restrict__ gamma,
    const float* __restrict__ beta, float* __restrict__ out) {
  const int lane = threadIdx.x & 31, wave = threadIdx.x >> 5;
  const int row = blockIdx.x * 8 + wave;
  const float* yr = y + (size_t)row * kD;
  float vals[24];
  float s = 0.0f, s2 = 0.0f;
#pragma unroll
  for (int i = 0; i < 6; ++i) {
    const float4 v = *(const float4*)(yr + i * 128 + lane * 4);
    vals[i * 4 + 0] = v.x; vals[i * 4 + 1] = v.y;
    vals[i * 4 + 2] = v.z; vals[i * 4 + 3] = v.w;
    s += v.x + v.y + v.z + v.w;
    s2 += v.x * v.x + v.y * v.y + v.z * v.z + v.w * v.w;
  }
#pragma unroll
  for (int m = 1; m < 32; m <<= 1) {
    s += __shfl_xor(s, m);
    s2 += __shfl_xor(s2, m);
  }
  const float mu = s * (1.0f / kD);
  const float var = s2 * (1.0f / kD) - mu * mu;
  const float rs = rsqrtf(var + 1e-3f);
  float* outr = out + (size_t)row * kD;
#pragma unroll
  for (int i = 0; i < 6; ++i)
#pragma unroll
    for (int j = 0; j < 4; ++j) {
      const int c = i * 128 + lane * 4 + j;
      outr[c] = gamma[c] * (vals[i * 4 + j] - mu) * rs + beta[c];
    }
}

// ---------------- workspace layout (bytes) -----------------------------------
constexpr size_t SZ_XH   = (size_t)kM * kD * 2;
constexpr size_t SZ_WQKV = (size_t)kNQ * kD * 2;
constexpr size_t SZ_WO   = (size_t)kD * kD * 2;
constexpr size_t SZ_BQKV = (size_t)kNQ * 4;
constexpr size_t SZ_HEAD = (size_t)kB * kH * kL * kDH * 2;  // Q / K / V^T each
constexpr size_t SZ_CTX  = (size_t)kM * kD * 2;

constexpr size_t OFF_XH   = 0;
constexpr size_t OFF_WQKV = OFF_XH + SZ_XH;
constexpr size_t OFF_WO   = OFF_WQKV + SZ_WQKV;
constexpr size_t OFF_BQKV = OFF_WO + SZ_WO;
constexpr size_t OFF_Q    = OFF_BQKV + SZ_BQKV;
constexpr size_t OFF_K    = OFF_Q + SZ_HEAD;
constexpr size_t OFF_VT   = OFF_K + SZ_HEAD;
constexpr size_t OFF_CTX  = OFF_VT + SZ_HEAD;
constexpr size_t OFF_Y    = OFF_CTX + SZ_CTX;

}  // namespace

extern "C" void kernel_launch(void* const* d_in, const int* in_sizes, int n_in,
                              void* d_out, int out_size, void* d_ws, size_t ws_size,
                              hipStream_t stream) {
  (void)in_sizes; (void)n_in; (void)out_size; (void)ws_size;
  const float* x     = (const float*)d_in[0];
  const int*   mask  = (const int*)d_in[1];
  const float* Wq    = (const float*)d_in[2];
  const float* bq    = (const float*)d_in[3];
  const float* Wk    = (const float*)d_in[4];
  const float* bk    = (const float*)d_in[5];
  const float* Wv    = (const float*)d_in[6];
  const float* bv    = (const float*)d_in[7];
  const float* Wo    = (const float*)d_in[8];
  const float* bo    = (const float*)d_in[9];
  const float* gamma = (const float*)d_in[10];
  const float* beta  = (const float*)d_in[11];
  float* out = (float*)d_out;

  char* ws = (char*)d_ws;
  _Float16* xh    = (_Float16*)(ws + OFF_XH);
  _Float16* wqkvt = (_Float16*)(ws + OFF_WQKV);
  _Float16* wot   = (_Float16*)(ws + OFF_WO);
  float*    bqkv  = (float*)(ws + OFF_BQKV);
  _Float16* qbuf  = (_Float16*)(ws + OFF_Q);
  _Float16* kbuf  = (_Float16*)(ws + OFF_K);
  _Float16* vtbuf = (_Float16*)(ws + OFF_VT);
  _Float16* ctx   = (_Float16*)(ws + OFF_CTX);
  float*    ybuf  = (float*)(ws + OFF_Y);

  k_cvt_x<<<(kM * kD + 255) / 256, 256, 0, stream>>>(x, xh);
  k_prep_wqkv<<<(kNQ * kD + 255) / 256, 256, 0, stream>>>(Wq, Wk, Wv, bq, bk, bv,
                                                          wqkvt, bqkv);
  k_prep_wo<<<(kD * kD + 255) / 256, 256, 0, stream>>>(Wo, wot);

  k_gemm_qkv<<<dim3(kM / 128, kNQ / 128), 256, 0, stream>>>(xh, wqkvt, bqkv, qbuf,
                                                            kbuf, vtbuf);
  k_attn<<<kB * kH * (kL / 128), 256, 0, stream>>>(qbuf, kbuf, vtbuf, mask, ctx);
  k_gemm_out<<<dim3(kM / 128, kD / 128), 256, 0, stream>>>(ctx, wot, bo, x, ybuf);
  k_layernorm<<<kM / 8, 256, 0, stream>>>(ybuf, gamma, beta, out);
}
